// SelfAttentionWithMemory_63591285784600
// MI455X (gfx1250) — matrix operand order repaired and checked by reading, not compiled, not run on hardware
//
#include <hip/hip_runtime.h>
#include <hip/hip_bf16.h>

typedef __attribute__((ext_vector_type(16))) __bf16       v16bf;
typedef __attribute__((ext_vector_type(8)))  float        v8f;
typedef __attribute__((ext_vector_type(4)))  unsigned int v4u;
typedef __attribute__((ext_vector_type(2)))  float        v2f;
typedef __attribute__((ext_vector_type(4)))  int          v4i;
typedef __attribute__((address_space(1)))    v4i*         gv4i_p;   // global int4*
typedef __attribute__((address_space(3)))    v4i*         lv4i_p;   // LDS int4*

#define HEADS   12
#define BATCH   64
#define SQ      198
#define SKV     206
#define DMODEL  768
#define DH      64
#define SP      208                  // padded token count (13*16)
#define QK_STRIDE (SP*DH)            // 13312 ushorts per (b,h) for Q and K
#define VT_ROW    224                // padded Vt row stride in ws (ushorts)
#define VT_STRIDE (DH*VT_ROW)        // 14336 ushorts per (b,h) for Vt
#define KOFF  ((size_t)BATCH*HEADS*QK_STRIDE)
#define VOFF  ((size_t)2*BATCH*HEADS*QK_STRIDE)

// kernel B dynamic-LDS partition (ushort offsets)
#define LK_ROW  72                   // K tile LDS row stride (halfs) - bank-conflict free
#define LV_ROW  232                  // Vt tile LDS row stride (halfs)
#define LP_ROW  232                  // P tile row stride (halfs)
#define LKU     0
#define LVU     (SP*LK_ROW)                 // 14976
#define LPU     (LVU + DH*LV_ROW)           // 29824
#define SMEM_USHORTS (LPU + 8*16*LP_ROW)    // 59520
#define SMEM_BYTES   (SMEM_USHORTS*2)       // 119040

#if defined(__HIP_DEVICE_COMPILE__) && __has_builtin(__builtin_amdgcn_global_load_async_to_lds_b128)
#define HAVE_ASYNC_LDS 1
#else
#define HAVE_ASYNC_LDS 0
#endif

__device__ __forceinline__ unsigned short f2bf(float x){
  union { __bf16 h; unsigned short u; } t;
  t.h = (__bf16)x;                       // native v_cvt f32->bf16
  return t.u;
}
__device__ __forceinline__ unsigned int pk2bf(float a, float b){
  union { __bf16 h[2]; unsigned int u; } t;
  t.h[0] = (__bf16)a; t.h[1] = (__bf16)b;
  return t.u;
}

union FragBF { v16bf v; v4u q[2]; };

// ---------------------------------------------------------------------------
// Kernel A: fused QKV projection. One block per (b, h). A = [hidden | memory]
// (206x768, 208 padded), W-slices are the 64 head columns of Wq/Wk/Wv.
// Output: Q,K row-major bf16 (208x64); V transposed bf16 (64 x VT_ROW).
// ---------------------------------------------------------------------------
__global__ __launch_bounds__(512, 1) void qkv_kernel(
    const float* __restrict__ hs, const float* __restrict__ mem,
    const float* __restrict__ Wq, const float* __restrict__ bq,
    const float* __restrict__ Wk, const float* __restrict__ bk,
    const float* __restrict__ Wv, const float* __restrict__ bv,
    unsigned short* __restrict__ ws)
{
  __shared__ __align__(16) unsigned int lA32[SP * 20];     // 208 rows x 32 halfs (pad 40)
  __shared__ __align__(16) unsigned int lW32[3][DH * 20];  // 64 n-rows x 32 halfs (pad 40)

  const int tid  = threadIdx.x;
  const int wave = tid >> 5;
  const int lane = tid & 31;
  const int ln   = lane & 15;
  const int hf   = lane >> 4;
  const int blk  = blockIdx.x;              // b*HEADS + h
  const int b    = blk / HEADS;
  const int h    = blk % HEADS;

  const unsigned short* lA = (const unsigned short*)lA32;

  v8f accQ[4] = {}, accK[4] = {}, accV[4] = {};
  const float* Wp[3] = {Wq, Wk, Wv};

  for (int kc = 0; kc < 24; ++kc) {
    const int k0 = kc * 32;
    // ---- stage A chunk: 2 fp32 -> packed bf16x2 -> ds_store_b32 ----
    #pragma unroll
    for (int i = 0; i < 7; ++i) {                   // 3328 dwords = 208*16
      int e = tid + 512 * i;
      if (e < 3328) {
        int row = e >> 4, cp = e & 15;
        float a0 = 0.f, a1 = 0.f;
        if (row < SQ) {
          v2f v = *(const v2f*)&hs[((size_t)b * SQ + row) * DMODEL + k0 + 2 * cp];
          a0 = v.x; a1 = v.y;
        } else if (row < SKV) {
          v2f v = *(const v2f*)&mem[(size_t)(row - SQ) * DMODEL + k0 + 2 * cp];
          a0 = v.x; a1 = v.y;
        }
        lA32[row * 20 + cp] = pk2bf(a0, a1);
      }
    }
    // ---- stage W chunks transposed to [n][k], packed pairs along k ----
    #pragma unroll
    for (int w = 0; w < 3; ++w) {
      const float* Wm = Wp[w];
      #pragma unroll
      for (int i = 0; i < 2; ++i) {                 // 1024 dwords = 16*64
        int e = tid + 512 * i;
        int kp = e >> 6, n = e & 63;
        size_t gi = (size_t)(k0 + 2 * kp) * DMODEL + h * DH + n;
        lW32[w][n * 20 + kp] = pk2bf(Wm[gi], Wm[gi + DMODEL]);
      }
    }
    if (kc + 1 < 24)   // global_prefetch_b8 for the next A chunk
      __builtin_prefetch(&hs[((size_t)b * SQ + (tid >> 1)) * DMODEL + k0 + 32], 0, 1);
    __syncthreads();

    // ---- 52 (mtile,ntile) pairs over 16 waves; 3 WMMAs per pair ----
    #pragma unroll
    for (int i = 0; i < 4; ++i) {
      int p = wave + 16 * i;
      if (p < 52) {
        int mt = p % 13, nt = p / 13;
        FragBF fa;
        const unsigned short* ap = &lA[(mt * 16 + ln) * 40 + hf * 16];
        fa.q[0] = *(const v4u*)ap;
        fa.q[1] = *(const v4u*)(ap + 8);
        FragBF fb;
        const unsigned short* w0 = (const unsigned short*)lW32[0] + (nt * 16 + ln) * 40 + hf * 16;
        fb.q[0] = *(const v4u*)w0; fb.q[1] = *(const v4u*)(w0 + 8);
        accQ[i] = __builtin_amdgcn_wmma_f32_16x16x32_bf16(false, fa.v, false, fb.v,
                                                          (short)0, accQ[i], false, false);
        const unsigned short* w1 = (const unsigned short*)lW32[1] + (nt * 16 + ln) * 40 + hf * 16;
        fb.q[0] = *(const v4u*)w1; fb.q[1] = *(const v4u*)(w1 + 8);
        accK[i] = __builtin_amdgcn_wmma_f32_16x16x32_bf16(false, fa.v, false, fb.v,
                                                          (short)0, accK[i], false, false);
        const unsigned short* w2 = (const unsigned short*)lW32[2] + (nt * 16 + ln) * 40 + hf * 16;
        fb.q[0] = *(const v4u*)w2; fb.q[1] = *(const v4u*)(w2 + 8);
        accV[i] = __builtin_amdgcn_wmma_f32_16x16x32_bf16(false, fa.v, false, fb.v,
                                                          (short)0, accV[i], false, false);
      }
    }
    __syncthreads();
  }

  // ---- epilogue: bias add, bf16 convert, store to workspace ----
  #pragma unroll
  for (int i = 0; i < 4; ++i) {
    int p = wave + 16 * i;
    if (p >= 52) continue;
    int mt = p % 13, nt = p / 13;
    float biq = bq[h * DH + nt * 16 + ln];
    float bik = bk[h * DH + nt * 16 + ln];
    float biv = bv[h * DH + nt * 16 + ln];
    unsigned short* qb = ws + (size_t)blk * QK_STRIDE;
    unsigned short* kb = ws + KOFF + (size_t)blk * QK_STRIDE;
    #pragma unroll
    for (int r = 0; r < 8; ++r) {
      int m = mt * 16 + r + 8 * hf;       // C/D layout: VGPR r -> row r (+8 for hi half)
      qb[m * DH + nt * 16 + ln] = f2bf(accQ[i][r] + biq);
      kb[m * DH + nt * 16 + ln] = f2bf(accK[i][r] + bik);
    }
    v4u o;
    o.x = pk2bf(accV[i][0] + biv, accV[i][1] + biv);
    o.y = pk2bf(accV[i][2] + biv, accV[i][3] + biv);
    o.z = pk2bf(accV[i][4] + biv, accV[i][5] + biv);
    o.w = pk2bf(accV[i][6] + biv, accV[i][7] + biv);
    unsigned short* vb = ws + VOFF + (size_t)blk * VT_STRIDE
                       + (nt * 16 + ln) * VT_ROW + mt * 16 + 8 * hf;
    *(v4u*)vb = o;                         // contiguous along token dim -> b128 store
  }
}

// ---------------------------------------------------------------------------
// Kernel B: attention per (b, h). K / Vt tiles async-DMAed into LDS once,
// then 13 Q row-tiles (8 waves) run entirely out of LDS. Scores via WMMA,
// analytic mask, in-register softmax (shfl_xor over 16-lane halves), P staged
// bf16 in per-wave LDS tile, ctx = P x V via WMMA.
// ---------------------------------------------------------------------------
__global__ __launch_bounds__(256, 1) void attn_kernel(
    const unsigned short* __restrict__ ws, float* __restrict__ out)
{
  extern __shared__ __align__(16) unsigned short smem[];
  unsigned short* lK = smem + LKU;          // 208 x 72
  unsigned short* lV = smem + LVU;          // 64  x 232 (224 used)
  const int tid  = threadIdx.x;
  const int wave = tid >> 5;
  const int lane = tid & 31;
  const int ln   = lane & 15;
  const int hf   = lane >> 4;
  const int bh   = blockIdx.x;
  const int b    = bh / HEADS;
  const int h    = bh % HEADS;
  unsigned short* lP = smem + LPU + wave * (16 * LP_ROW);

  const unsigned short* Qb = ws + (size_t)bh * QK_STRIDE;
  const unsigned short* Kb = ws + KOFF + (size_t)bh * QK_STRIDE;
  const unsigned short* Vb = ws + VOFF + (size_t)bh * VT_STRIDE;

#if HAVE_ASYNC_LDS
  // ---- async global->LDS DMA of K (1664 x 16B) and Vt (1792 x 16B) ----
  #pragma unroll
  for (int i = 0; i < 7; ++i) {
    int c = tid + 256 * i;
    if (c < 1664) {                         // 208 rows * 8 chunks
      int row = c >> 3, off = c & 7;
      __builtin_amdgcn_global_load_async_to_lds_b128(
          (gv4i_p)(Kb + row * DH + off * 8),
          (lv4i_p)(lK + row * LK_ROW + off * 8), 0, 0);
    }
  }
  #pragma unroll
  for (int i = 0; i < 7; ++i) {
    int c = tid + 256 * i;                  // 64 rows * 28 chunks == 1792 = 7*256
    int row = c / 28, off = c % 28;
    __builtin_amdgcn_global_load_async_to_lds_b128(
        (gv4i_p)(Vb + row * VT_ROW + off * 8),
        (lv4i_p)(lV + row * LV_ROW + off * 8), 0, 0);
  }
#else
  // ---- fallback: plain staged copies ----
  {
    const unsigned int* Kg = (const unsigned int*)Kb;
    unsigned int* lK32 = (unsigned int*)lK;
    #pragma unroll
    for (int i = 0; i < 26; ++i) {          // 6656 dwords
      int e = tid + 256 * i;
      int row = e >> 5, cu = e & 31;
      lK32[row * (LK_ROW / 2) + cu] = Kg[e];
    }
    const unsigned int* Vg = (const unsigned int*)Vb;
    unsigned int* lV32 = (unsigned int*)lV;
    #pragma unroll
    for (int i = 0; i < 28; ++i) {          // 7168 dwords
      int e = tid + 256 * i;
      int row = e / 112, cu = e % 112;
      lV32[row * (LV_ROW / 2) + cu] = Vg[e];
    }
  }
#endif

  // zero own P tile while the DMA is in flight (pad cols must be 0 for PV tail)
  {
    unsigned int* pz = (unsigned int*)lP;
    for (int i = lane; i < 16 * LP_ROW / 2; i += 32) pz[i] = 0u;
  }
#if HAVE_ASYNC_LDS
  asm volatile("s_wait_asynccnt 0x0" ::: "memory");
#endif
  asm volatile("s_wait_dscnt 0" ::: "memory");
  __syncthreads();

  for (int iter = 0; iter < 2; ++iter) {
    int qt = wave + 8 * iter;
    if (qt >= 13) break;                    // wave-uniform
    int q0 = qt * 16;

    // Q A-fragments for both 32-wide k-steps of dh=64 (global, L2-resident)
    FragBF fq[2];
    #pragma unroll
    for (int s = 0; s < 2; ++s) {
      const unsigned short* qp = Qb + (q0 + ln) * DH + s * 32 + hf * 16;
      fq[s].q[0] = *(const v4u*)qp;
      fq[s].q[1] = *(const v4u*)(qp + 8);
    }

    // ---- scores S = (Q K^T) * scale + mask ----
    v8f sc[13] = {};
    #pragma unroll
    for (int t = 0; t < 13; ++t) {
      #pragma unroll
      for (int s = 0; s < 2; ++s) {
        FragBF fk;
        const unsigned short* kp = lK + (t * 16 + ln) * LK_ROW + s * 32 + hf * 16;
        fk.q[0] = *(const v4u*)kp; fk.q[1] = *(const v4u*)(kp + 8);
        sc[t] = __builtin_amdgcn_wmma_f32_16x16x32_bf16(false, fq[s].v, false, fk.v,
                                                        (short)0, sc[t], false, false);
      }
      int col = t * 16 + ln;
      #pragma unroll
      for (int r = 0; r < 8; ++r) {
        int q = q0 + r + 8 * hf;
        float v = sc[t][r] * 0.125f;        // 1/sqrt(64)
        bool ok = (col < 197) || (q == 197 && col < 206);
        sc[t][r] = ok ? v : -1e30f;
      }
    }

    // ---- softmax: each row lives in one 16-lane half at fixed acc index ----
    float sm[8];
    #pragma unroll
    for (int r = 0; r < 8; ++r) {
      float m = -1e30f;
      #pragma unroll
      for (int t = 0; t < 13; ++t) m = fmaxf(m, sc[t][r]);
      #pragma unroll
      for (int off = 8; off >= 1; off >>= 1) m = fmaxf(m, __shfl_xor(m, off, 16));
      float s = 0.f;
      #pragma unroll
      for (int t = 0; t < 13; ++t) { float p = __expf(sc[t][r] - m); sc[t][r] = p; s += p; }
      #pragma unroll
      for (int off = 8; off >= 1; off >>= 1) s += __shfl_xor(s, off, 16);
      sm[r] = s;
    }

    // ---- stage P (bf16) into this wave's LDS tile ----
    #pragma unroll
    for (int t = 0; t < 13; ++t) {
      int col = t * 16 + ln;
      #pragma unroll
      for (int r = 0; r < 8; ++r)
        lP[(r + 8 * hf) * LP_ROW + col] = f2bf(sc[t][r]);
    }
    asm volatile("s_wait_dscnt 0" ::: "memory");

    // ---- ctx = P @ V (7 k-steps over 224 padded token cols) ----
    #pragma unroll
    for (int j = 0; j < 4; ++j) {
      v8f c = {};
      #pragma unroll
      for (int s = 0; s < 7; ++s) {
        FragBF fp, fv;
        const unsigned short* pp = lP + ln * LP_ROW + s * 32 + hf * 16;
        fp.q[0] = *(const v4u*)pp; fp.q[1] = *(const v4u*)(pp + 8);
        const unsigned short* vp = lV + (j * 16 + ln) * LV_ROW + s * 32 + hf * 16;
        fv.q[0] = *(const v4u*)vp; fv.q[1] = *(const v4u*)(vp + 8);
        c = __builtin_amdgcn_wmma_f32_16x16x32_bf16(false, fp.v, false, fv.v,
                                                    (short)0, c, false, false);
      }
      #pragma unroll
      for (int r = 0; r < 8; ++r) {
        int q = q0 + r + 8 * hf;
        if (q < SQ)
          out[((size_t)b * SQ + q) * DMODEL + h * DH + j * 16 + ln] = c[r] / sm[r];
      }
    }
  }
}

extern "C" void kernel_launch(void* const* d_in, const int* in_sizes, int n_in,
                              void* d_out, int out_size, void* d_ws, size_t ws_size,
                              hipStream_t stream) {
  const float* hs  = (const float*)d_in[0];
  const float* mem = (const float*)d_in[1];
  const float* Wq  = (const float*)d_in[2];
  const float* bq  = (const float*)d_in[3];
  const float* Wk  = (const float*)d_in[4];
  const float* bk  = (const float*)d_in[5];
  const float* Wv  = (const float*)d_in[6];
  const float* bv  = (const float*)d_in[7];
  unsigned short* ws = (unsigned short*)d_ws;
  float* out = (float*)d_out;

  (void)hipFuncSetAttribute((const void*)attn_kernel,
                            hipFuncAttributeMaxDynamicSharedMemorySize, SMEM_BYTES);

  qkv_kernel<<<dim3(BATCH * HEADS), dim3(512), 0, stream>>>(hs, mem, Wq, bq, Wk, bk, Wv, bv, ws);
  attn_kernel<<<dim3(BATCH * HEADS), dim3(256), SMEM_BYTES, stream>>>(ws, out);
}
